// CornerKickPredictor_1219770712592
// MI455X (gfx1250) — matrix operand order repaired and checked by reading, not compiled, not run on hardware
//
#include <hip/hip_runtime.h>
#include <hip/hip_bf16.h>

typedef __attribute__((ext_vector_type(16))) _Float16 v16h;
typedef __attribute__((ext_vector_type(8)))  float    v8f;

#define NNODES 23
#define NHEADS 4
#define MAXE   536   // padded edge capacity (506 + 23 self loops = 529)

// ---------------------------------------------------------------------------
// Build target-sorted edge list (CSR) once: edge_index is identical for all
// 1024 graphs, so softmax segments become contiguous ranges (no atomics).
// ---------------------------------------------------------------------------
__global__ void build_csr_kernel(const int* __restrict__ ei, int E,
                                 int* __restrict__ csr_ptr,
                                 int* __restrict__ es_src,
                                 int* __restrict__ es_tgt) {
  if (threadIdx.x != 0 || blockIdx.x != 0) return;
  int cnt[NNODES];
  for (int n = 0; n < NNODES; ++n) cnt[n] = 0;
  const int Etot = E + NNODES;
  for (int e = 0; e < Etot; ++e) {
    int tgt = (e < E) ? ei[E + e] : (e - E);   // self loops appended
    cnt[tgt]++;
  }
  int ptr[NNODES + 1];
  ptr[0] = 0;
  for (int n = 0; n < NNODES; ++n) ptr[n + 1] = ptr[n] + cnt[n];
  for (int n = 0; n <= NNODES; ++n) csr_ptr[n] = ptr[n];
  int pos[NNODES];
  for (int n = 0; n < NNODES; ++n) pos[n] = ptr[n];
  for (int e = 0; e < Etot; ++e) {
    int src = (e < E) ? ei[e]     : (e - E);
    int tgt = (e < E) ? ei[E + e] : (e - E);
    int j = pos[tgt]++;
    es_src[j] = src;
    es_tgt[j] = tgt;
  }
}

// ---------------------------------------------------------------------------
// Fused GATv2 layer. One 256-thread block (8 waves) per graph.
//   C = per-head channels (64 layer 1, 32 layer 2)
//   K = input feature dim  (8 layer 1, 64 layer 2)
// WMMA operand paths are fully branchless: A comes from a zero-padded LDS
// tile, B uses clamped-index loads + value select, C/D stores go through a
// row-clamped junk row. EXEC stays all-ones across the WMMA region.
// ---------------------------------------------------------------------------
template<int C, int K>
__global__ void __launch_bounds__(256)
gat_layer_kernel(const float* __restrict__ xin,
                 const float* __restrict__ Wl,
                 const float* __restrict__ Wr,
                 const float* __restrict__ aW,     // (4, C)
                 const float* __restrict__ bias,   // (C,)
                 const float* __restrict__ gamma,  // (C,)
                 const float* __restrict__ beta,   // (C,)
                 const int*  __restrict__ csr_ptr,
                 const int*  __restrict__ es_src,
                 const int*  __restrict__ es_tgt,
                 int Etot,
                 float* __restrict__ h_out,        // (G, N, C) or nullptr
                 float* __restrict__ emb) {        // (G, C)    or nullptr
  constexpr int HC = NHEADS * C;
  constexpr int KP = (K + 31) & ~31;               // K padded to WMMA depth
  constexpr int SCR = (32 * KP > NNODES * C) ? 32 * KP : NNODES * C;

  __shared__ float scratch_s[SCR];                 // padded A tile, then pre-LN
  __shared__ float xl_s[24 * HC];                  // row 23 = write junk
  __shared__ float xr_s[24 * HC];                  // later reused as accumulator
  __shared__ float logit_s[MAXE * NHEADS];         // later holds alpha
  __shared__ float m_s[NNODES * NHEADS];           // later: LN mean
  __shared__ float d_s[NNODES * NHEADS];           // later: LN rstd

  const int g   = blockIdx.x;
  const int tid = threadIdx.x;
  const float* xg = xin + (size_t)g * NNODES * K;

  // ---- stage node features into zero-padded [32 x KP] tile ----
  for (int i = tid; i < 32 * KP; i += 256) scratch_s[i] = 0.0f;
  __syncthreads();
  for (int i = tid; i < NNODES * K; i += 256)
    scratch_s[(i / K) * KP + (i % K)] = xg[i];
  __syncthreads();

  // ---- projections via WMMA f32<-f16 16x16x32 (branchless operand paths) ----
  {
    const int wave = tid >> 5;
    const int lane = tid & 31;
    const int lg   = lane >> 4;    // lane group 0/1
    const int ln   = lane & 15;
    constexpr int NT    = HC / 16;
    constexpr int TOTAL = 2 * 2 * NT;        // {l,r} x 2 M-tiles x N-tiles
    for (int job = wave; job < TOTAL; job += 8) {
      const int lr = job & 1;
      const int mt = (job >> 1) & 1;
      const int nt = job >> 2;
      const float* W  = lr ? Wr : Wl;
      float*      dst = lr ? xr_s : xl_s;
      const int arow = mt * 16 + ln;
      const int col  = nt * 16 + ln;
      v8f acc = {};
      for (int kk = 0; kk < K; kk += 32) {
        v16h af, bf;
        #pragma unroll
        for (int i = 0; i < 16; ++i) {        // A 16x32 f16 layout (padded LDS)
          const int j2 = i >> 1, t = i & 1;
          const int kl = (j2 < 4) ? (2 * j2 + 8 * lg + t)
                                  : (16 + 2 * (j2 - 4) + 8 * lg + t);
          af[i] = (_Float16)scratch_s[arow * KP + kk + kl];
        }
        #pragma unroll
        for (int i = 0; i < 16; ++i) {        // B 32x16 f16 layout
          const int k = kk + ((i < 8) ? (i + 8 * lg) : (16 + 8 * lg + (i - 8)));
          float v;
          if constexpr (K % 32 == 0) {
            v = W[k * HC + col];              // always in range
          } else {
            const int kc = (k < K) ? k : 0;   // clamped addr, selected value
            v = W[kc * HC + col];
            v = (k < K) ? v : 0.0f;
          }
          bf[i] = (_Float16)v;
        }
        acc = __builtin_amdgcn_wmma_f32_16x16x32_f16(false, af, false, bf,
                                                     (short)0, acc, false, false);
      }
      #pragma unroll
      for (int r = 0; r < 8; ++r) {           // C/D 16x16 f32 layout
        const int row = mt * 16 + r + 8 * lg;
        const int rowc = (row < NNODES) ? row : NNODES;   // junk row 23
        dst[rowc * HC + col] = acc[r];
      }
    }
  }
  __syncthreads();

  // ---- per-(edge, head) logits: a . leaky_relu(xl[src]+xr[tgt]) ----
  for (int idx = tid; idx < Etot * NHEADS; idx += 256) {
    const int j = idx >> 2;
    const int h = idx & 3;
    const int s = es_src[j];
    const int t = es_tgt[j];
    const float* pl = xl_s + s * HC + h * C;
    const float* pr = xr_s + t * HC + h * C;
    const float* pa = aW + h * C;
    float acc = 0.0f;
    for (int c = 0; c < C; ++c) {
      float u = pl[c] + pr[c];
      u = (u > 0.0f) ? u : 0.2f * u;
      acc += u * pa[c];
    }
    logit_s[idx] = acc;
  }
  __syncthreads();

  // ---- segment max & sum-of-exp per (target, head): contiguous ranges ----
  for (int idx = tid; idx < NNODES * NHEADS; idx += 256) {
    const int n = idx >> 2;
    const int h = idx & 3;
    const int r0 = csr_ptr[n], r1 = csr_ptr[n + 1];
    float m = -3.0e38f;
    for (int j = r0; j < r1; ++j) m = fmaxf(m, logit_s[j * NHEADS + h]);
    float s = 0.0f;
    for (int j = r0; j < r1; ++j) s += expf(logit_s[j * NHEADS + h] - m);
    m_s[idx] = m;
    d_s[idx] = s;
  }
  __syncthreads();

  // ---- alpha (overwrite logits) ----
  for (int idx = tid; idx < Etot * NHEADS; idx += 256) {
    const int j = idx >> 2;
    const int h = idx & 3;
    const int t = es_tgt[j];
    logit_s[idx] = expf(logit_s[idx] - m_s[t * NHEADS + h]) / d_s[t * NHEADS + h];
  }
  __syncthreads();

  // ---- aggregate messages: out[n][hc] = sum_e alpha[e][h] * xl[src][hc] ----
  // (xr_s recycled as accumulator; every element fully written)
  for (int item = tid; item < NNODES * HC; item += 256) {
    const int n  = item / HC;
    const int hc = item % HC;
    const int h  = hc / C;
    const int r0 = csr_ptr[n], r1 = csr_ptr[n + 1];
    float acc = 0.0f;
    for (int j = r0; j < r1; ++j)
      acc += logit_s[j * NHEADS + h] * xl_s[es_src[j] * HC + hc];
    xr_s[item] = acc;
  }
  __syncthreads();

  // ---- mean over heads + bias (pre_s aliases the dead padded-A tile) ----
  float* pre_s = scratch_s;
  for (int item = tid; item < NNODES * C; item += 256) {
    const int n = item / C;
    const int c = item % C;
    float s = 0.0f;
    #pragma unroll
    for (int h = 0; h < NHEADS; ++h) s += xr_s[n * HC + h * C + c];
    pre_s[item] = s * (1.0f / NHEADS) + bias[c];
  }
  __syncthreads();

  // ---- LayerNorm stats (one thread per node) ----
  if (tid < NNODES) {
    float mu = 0.0f;
    for (int c = 0; c < C; ++c) mu += pre_s[tid * C + c];
    mu *= (1.0f / C);
    float var = 0.0f;
    for (int c = 0; c < C; ++c) { float d = pre_s[tid * C + c] - mu; var += d * d; }
    var *= (1.0f / C);
    m_s[tid] = mu;
    d_s[tid] = rsqrtf(var + 1e-5f);
  }
  __syncthreads();

  // ---- normalize + affine + ReLU; write h_out ----
  for (int item = tid; item < NNODES * C; item += 256) {
    const int n = item / C;
    const int c = item % C;
    float v = (pre_s[item] - m_s[n]) * d_s[n] * gamma[c] + beta[c];
    v = fmaxf(v, 0.0f);
    pre_s[item] = v;
    if (h_out) h_out[((size_t)g * NNODES + n) * C + c] = v;
  }
  if (emb) {
    __syncthreads();
    for (int c = tid; c < C; c += 256) {
      float s = 0.0f;
      for (int n = 0; n < NNODES; ++n) s += pre_s[n * C + c];
      emb[(size_t)g * C + c] = s * (1.0f / NNODES);
    }
  }
}

// ---------------------------------------------------------------------------
// Both GRU layers + all 4 heads, one block per batch row (B=16). ys0 kept in
// LDS; hT feeds the heads directly.
// ---------------------------------------------------------------------------
__global__ void __launch_bounds__(192)
gru_heads_kernel(const float* __restrict__ emb,   // (B*T, 32), g = b*T + t
                 const float* __restrict__ Wx0, const float* __restrict__ Wh0,
                 const float* __restrict__ bx0, const float* __restrict__ bh0,
                 const float* __restrict__ Wx1, const float* __restrict__ Wh1,
                 const float* __restrict__ bx1, const float* __restrict__ bh1,
                 const float* __restrict__ Ws,  const float* __restrict__ bs,
                 const float* __restrict__ Wg,  const float* __restrict__ bg,
                 const float* __restrict__ Wc,  const float* __restrict__ bc,
                 const float* __restrict__ Wo,  const float* __restrict__ bo,
                 float* __restrict__ out) {
  constexpr int T = 64, HD = 64, CI = 32, G3 = 192;
  __shared__ float ys0[T * HD];     // 16 KB
  __shared__ float h[HD];
  __shared__ float xb[CI];
  __shared__ float gxs[G3], ghs[G3];
  const int b   = blockIdx.x;
  const int tid = threadIdx.x;

  // ---- layer 0 ----
  if (tid < HD) h[tid] = 0.0f;
  __syncthreads();
  for (int t = 0; t < T; ++t) {
    if (tid < CI) xb[tid] = emb[((size_t)b * T + t) * CI + tid];
    __syncthreads();
    {
      float gx = bx0[tid];
      for (int k = 0; k < CI; ++k) gx += xb[k] * Wx0[k * G3 + tid];
      float gh = bh0[tid];
      for (int k = 0; k < HD; ++k) gh += h[k] * Wh0[k * G3 + tid];
      gxs[tid] = gx; ghs[tid] = gh;
    }
    __syncthreads();
    if (tid < HD) {
      float r  = 1.0f / (1.0f + expf(-(gxs[tid] + ghs[tid])));
      float z  = 1.0f / (1.0f + expf(-(gxs[HD + tid] + ghs[HD + tid])));
      float nn = tanhf(gxs[2 * HD + tid] + r * ghs[2 * HD + tid]);
      float hn = (1.0f - z) * nn + z * h[tid];
      h[tid] = hn;
      ys0[t * HD + tid] = hn;
    }
    __syncthreads();
  }

  // ---- layer 1 ----
  if (tid < HD) h[tid] = 0.0f;
  __syncthreads();
  for (int t = 0; t < T; ++t) {
    {
      float gx = bx1[tid];
      for (int k = 0; k < HD; ++k) gx += ys0[t * HD + k] * Wx1[k * G3 + tid];
      float gh = bh1[tid];
      for (int k = 0; k < HD; ++k) gh += h[k] * Wh1[k * G3 + tid];
      gxs[tid] = gx; ghs[tid] = gh;
    }
    __syncthreads();
    if (tid < HD) {
      float r  = 1.0f / (1.0f + expf(-(gxs[tid] + ghs[tid])));
      float z  = 1.0f / (1.0f + expf(-(gxs[HD + tid] + ghs[HD + tid])));
      float nn = tanhf(gxs[2 * HD + tid] + r * ghs[2 * HD + tid]);
      h[tid] = (1.0f - z) * nn + z * h[tid];
    }
    __syncthreads();
  }

  // ---- heads: out = [shot(16) | goal(16) | contact(16x2) | outcome(16x6)] ----
  if (tid < 10) {
    float acc = 0.0f;
    if (tid == 0) {
      for (int k = 0; k < HD; ++k) acc += h[k] * Ws[k];
      out[b] = 1.0f / (1.0f + expf(-(acc + bs[0])));
    } else if (tid == 1) {
      for (int k = 0; k < HD; ++k) acc += h[k] * Wg[k];
      out[16 + b] = 1.0f / (1.0f + expf(-(acc + bg[0])));
    } else if (tid < 4) {
      int j = tid - 2;
      for (int k = 0; k < HD; ++k) acc += h[k] * Wc[k * 2 + j];
      out[32 + b * 2 + j] = acc + bc[j];
    } else {
      int j = tid - 4;
      for (int k = 0; k < HD; ++k) acc += h[k] * Wo[k * 6 + j];
      out[64 + b * 6 + j] = acc + bo[j];
    }
  }
}

// ---------------------------------------------------------------------------
extern "C" void kernel_launch(void* const* d_in, const int* in_sizes, int n_in,
                              void* d_out, int out_size, void* d_ws, size_t ws_size,
                              hipStream_t stream) {
  const float* x     = (const float*)d_in[0];
  const int*   ei    = (const int*)d_in[1];
  const float* Wl1   = (const float*)d_in[2];
  const float* Wr1   = (const float*)d_in[3];
  const float* a1    = (const float*)d_in[4];
  const float* bias1 = (const float*)d_in[5];
  const float* g1    = (const float*)d_in[6];
  const float* b1    = (const float*)d_in[7];
  const float* Wl2   = (const float*)d_in[8];
  const float* Wr2   = (const float*)d_in[9];
  const float* a2    = (const float*)d_in[10];
  const float* bias2 = (const float*)d_in[11];
  const float* g2    = (const float*)d_in[12];
  const float* b2    = (const float*)d_in[13];
  const float* Wx0   = (const float*)d_in[14];
  const float* Wh0   = (const float*)d_in[15];
  const float* bx0   = (const float*)d_in[16];
  const float* bh0   = (const float*)d_in[17];
  const float* Wx1   = (const float*)d_in[18];
  const float* Wh1   = (const float*)d_in[19];
  const float* bx1   = (const float*)d_in[20];
  const float* bh1   = (const float*)d_in[21];
  const float* Wsp   = (const float*)d_in[22];
  const float* bsp   = (const float*)d_in[23];
  const float* Wgp   = (const float*)d_in[24];
  const float* bgp   = (const float*)d_in[25];
  const float* Wcp   = (const float*)d_in[26];
  const float* bcp   = (const float*)d_in[27];
  const float* Wop   = (const float*)d_in[28];
  const float* bop   = (const float*)d_in[29];

  const int E = in_sizes[1] / 2;           // 506
  const int G = 1024;                      // B*T = 16*64

  // scratch layout
  float* ws  = (float*)d_ws;
  float* h1  = ws;                                   // G*23*64 floats
  float* emb = h1 + (size_t)G * NNODES * 64;         // G*32 floats
  int* csr_ptr = (int*)(emb + (size_t)G * 32);       // 24 ints (pad 32)
  int* es_src  = csr_ptr + 32;                       // MAXE ints
  int* es_tgt  = es_src + MAXE;                      // MAXE ints

  build_csr_kernel<<<1, 32, 0, stream>>>(ei, E, csr_ptr, es_src, es_tgt);

  gat_layer_kernel<64, 8><<<G, 256, 0, stream>>>(
      x, Wl1, Wr1, a1, bias1, g1, b1,
      csr_ptr, es_src, es_tgt, E + NNODES, h1, nullptr);

  gat_layer_kernel<32, 64><<<G, 256, 0, stream>>>(
      h1, Wl2, Wr2, a2, bias2, g2, b2,
      csr_ptr, es_src, es_tgt, E + NNODES, nullptr, emb);

  gru_heads_kernel<<<16, 192, 0, stream>>>(
      emb, Wx0, Wh0, bx0, bh0, Wx1, Wh1, bx1, bh1,
      Wsp, bsp, Wgp, bgp, Wcp, bcp, Wop, bop, (float*)d_out);
}
